// FourierPositionalBias_23381801959705
// MI455X (gfx1250) — matrix operand order repaired and checked by reading, not compiled, not run on hardware
//
#include <hip/hip_runtime.h>

// Problem constants (match reference)
#define NBATCH 4
#define NPOS   1024
#define NF     64
#define SCALE  10.0f

// Tiling
#define TM   128          // output rows per workgroup
#define TN   64           // output cols per workgroup
#define KD   128          // GEMM K = 2*NF  (cos-half | sin-half)
#define LDK  136          // padded LDS K-stride in halves (16B-aligned rows, bank-shifted)
#define NTHREADS 256      // 8 wave32

typedef _Float16 half_t;
typedef __attribute__((ext_vector_type(16))) _Float16 v16h;
typedef __attribute__((ext_vector_type(8)))  _Float16 v8h;
typedef __attribute__((ext_vector_type(8)))  float    v8f;

__global__ __launch_bounds__(NTHREADS)
void fourier_bias_wmma_kernel(const float* __restrict__ frac,   // [NBATCH][NPOS]
                              const float* __restrict__ W,      // [NF]
                              const float* __restrict__ bvec,   // [NF]
                              const float* __restrict__ amp,    // [NF]
                              float* __restrict__ out)          // [NBATCH][NPOS][NPOS]
{
    __shared__ half_t sA[TM * LDK];   // A[i][k]: k<64 -> a_f*cos(10 x_i w_f); k>=64 -> a_f*sin(...)
    __shared__ half_t sB[TN * LDK];   // B[j][k]: k<64 -> cos(10 x_j w_f + b_f); k>=64 -> sin(...)
    __shared__ float  sW[NF], sBias[NF], sAmp[NF];

    const int tid   = threadIdx.x;
    const int batch = blockIdx.z;
    const int i0    = blockIdx.y * TM;
    const int j0    = blockIdx.x * TN;

    if (tid < NF) {
        sW[tid]    = W[tid];
        sBias[tid] = bvec[tid];
        sAmp[tid]  = amp[tid];
    }
    __syncthreads();

    const float* fb = frac + batch * NPOS;

    // ---- Phase 1: build K-panels in LDS (tiny trig cost: 12K sincos / workgroup) ----
    // A-side: 128 rows x 64 features
    #pragma unroll
    for (int p = 0; p < (TM * NF) / NTHREADS; ++p) {        // 32 iters
        int idx = (p << 8) + tid;
        int row = idx >> 6;
        int f   = idx & (NF - 1);
        float th = SCALE * fb[i0 + row] * sW[f];
        float s, c;
        __sincosf(th, &s, &c);
        sA[row * LDK + f]      = (half_t)(sAmp[f] * c);
        sA[row * LDK + NF + f] = (half_t)(sAmp[f] * s);
    }
    // B-side: 64 cols x 64 features
    #pragma unroll
    for (int p = 0; p < (TN * NF) / NTHREADS; ++p) {        // 16 iters
        int idx = (p << 8) + tid;
        int row = idx >> 6;
        int f   = idx & (NF - 1);
        float th = SCALE * fb[j0 + row] * sW[f] + sBias[f];
        float s, c;
        __sincosf(th, &s, &c);
        sB[row * LDK + f]      = (half_t)c;
        sB[row * LDK + NF + f] = (half_t)s;
    }
    __syncthreads();

    // ---- Phase 2: WMMA GEMM. 8 waves; each owns a 32x32 sub-tile (2x2 accumulators) ----
    const int wid  = tid >> 5;           // wave id 0..7
    const int lane = tid & 31;
    const int wm   = wid & 3;            // 4 row-blocks of 32
    const int wn   = wid >> 2;           // 2 col-blocks of 32
    const int rbase = wm * 32;
    const int cbase = wn * 32;

    v8f acc[2][2] = {};

    // A fragment addressing (16-bit A 16x32 layout): lane<16 -> K {0..7,16..23}, lane>=16 -> K {8..15,24..31}
    const int arow = rbase + (lane & 15);
    const int akb  = (lane >> 4) * 8;
    // B fragment addressing (16-bit B 32x16 layout): lane n<16 -> K {0..7,16..23} of column n; lane>=16 -> K {8..15,24..31}
    const int bcol = cbase + (lane & 15);
    const int bkb  = (lane >> 4) * 8;

    #pragma unroll
    for (int ki = 0; ki < KD / 32; ++ki) {
        const int k0 = ki * 32;

        v16h af[2];
        #pragma unroll
        for (int mi = 0; mi < 2; ++mi) {
            const half_t* pa = &sA[(arow + 16 * mi) * LDK + k0 + akb];
            v8h lo = *(const v8h*)(pa);          // K = akb .. akb+7
            v8h hi = *(const v8h*)(pa + 16);     // K = akb+16 .. akb+23
            af[mi] = __builtin_shufflevector(lo, hi,
                     0,1,2,3,4,5,6,7,8,9,10,11,12,13,14,15);
        }
        v16h bf[2];
        #pragma unroll
        for (int ni = 0; ni < 2; ++ni) {
            const half_t* pb = &sB[(bcol + 16 * ni) * LDK + k0 + bkb];
            v8h lo = *(const v8h*)(pb);          // K = bkb .. bkb+7
            v8h hi = *(const v8h*)(pb + 16);     // K = bkb+16 .. bkb+23
            bf[ni] = __builtin_shufflevector(lo, hi,
                     0,1,2,3,4,5,6,7,8,9,10,11,12,13,14,15);
        }

        #pragma unroll
        for (int mi = 0; mi < 2; ++mi)
            #pragma unroll
            for (int ni = 0; ni < 2; ++ni)
                acc[mi][ni] = __builtin_amdgcn_wmma_f32_16x16x32_f16(
                    /*neg_a=*/false, af[mi], /*neg_b=*/false, bf[ni],
                    /*c_mod=*/(short)0, acc[mi][ni],
                    /*reuse_a=*/false, /*reuse_b=*/false);
    }

    // ---- Store: 16x16 f32 C/D layout: VGPR r -> M=r (lanes 0-15) / M=r+8 (lanes 16-31); N = lane&15 ----
    const int rlane = lane & 15;
    const int rhi   = (lane >> 4) << 3;   // 0 or 8
    float* ob = out + (size_t)batch * NPOS * NPOS;
    #pragma unroll
    for (int mi = 0; mi < 2; ++mi) {
        #pragma unroll
        for (int ni = 0; ni < 2; ++ni) {
            const int bi = i0 + rbase + 16 * mi + rhi;
            const int bj = j0 + cbase + 16 * ni + rlane;
            #pragma unroll
            for (int r = 0; r < 8; ++r)
                ob[(size_t)(bi + r) * NPOS + bj] = acc[mi][ni][r];
        }
    }
}

extern "C" void kernel_launch(void* const* d_in, const int* in_sizes, int n_in,
                              void* d_out, int out_size, void* d_ws, size_t ws_size,
                              hipStream_t stream) {
    (void)in_sizes; (void)n_in; (void)d_ws; (void)ws_size; (void)out_size;
    const float* frac = (const float*)d_in[0];   // [4,1024]
    const float* W    = (const float*)d_in[1];   // [1,64]
    const float* bvec = (const float*)d_in[2];   // [64]
    const float* amp  = (const float*)d_in[3];   // [64]
    float* out = (float*)d_out;                  // [4,1024,1024]

    dim3 grid(NPOS / TN, NPOS / TM, NBATCH);     // (16, 8, 4)
    fourier_bias_wmma_kernel<<<grid, NTHREADS, 0, stream>>>(frac, W, bvec, amp, out);
}
